// DenseNonEquivariantPointAttention_3685081940024
// MI455X (gfx1250) — compile-verified, matrix operand order checked by reading, not run on hardware
//
#include <hip/hip_runtime.h>
#include <hip/hip_bf16.h>
#include <math.h>

#define NTOK 512
#define DIM  256
#define SS   32
#define HH   4
#define PAIRC 56
#define P3   96
#define FDIM 112
#define HSX  128
#define RAWW 672

typedef __attribute__((ext_vector_type(16))) _Float16 v16h;
typedef __attribute__((ext_vector_type(8)))  _Float16 v8h;
typedef __attribute__((ext_vector_type(8)))  float    v8f;

// CDNA5 16-bit fragment K map (ISA 7.12.2): half t of lane (hi = lane>>4)
// holds K = 16*(t>>3) + 8*hi + (t&7)
__device__ __forceinline__ int frag_k(int t, int hi) {
  return ((t >> 3) << 4) + (hi << 3) + (t & 7);
}

// A fragment from a row-major f16 row: two contiguous 8-half runs -> 2x b128
__device__ __forceinline__ v16h frag_row(const _Float16* p, int hi) {
  v8h x = *(const v8h*)(p + hi * 8);
  v8h y = *(const v8h*)(p + 16 + hi * 8);
  v16h r;
#pragma unroll
  for (int t = 0; t < 8; ++t) { r[t] = x[t]; r[t + 8] = y[t]; }
  return r;
}

// B fragment from fragment-major packed buffer: 16 contiguous halves per lane
__device__ __forceinline__ v16h frag_packed(const _Float16* p) {
  v8h x = *(const v8h*)p;
  v8h y = *(const v8h*)(p + 8);
  v16h r;
#pragma unroll
  for (int t = 0; t < 8; ++t) { r[t] = x[t]; r[t + 8] = y[t]; }
  return r;
}

__device__ __forceinline__ v8f wmma_f16(v16h a, v16h b, v8f c) {
  return __builtin_amdgcn_wmma_f32_16x16x32_f16(false, a, false, b, (short)0, c,
                                                false, false);
}

__device__ __forceinline__ float wave_sum(float v) {
#pragma unroll
  for (int o = 16; o > 0; o >>= 1) v += __shfl_xor(v, o, 32);
  return v;
}

// ------------------------------------------------ K0a: misc (bias concat, PS)
__global__ void k0_misc(const float* b_q, const float* b_k, const float* b_v,
                        const float* b_qp, const float* b_kp, const float* b_vp,
                        const float* gamma, float* BCAT, float* PS) {
  int tid = threadIdx.x;
  for (int t = tid; t < RAWW; t += blockDim.x) {
    float b;
    if (t < 128) b = b_q[t];
    else if (t < 256) b = b_k[t - 128];
    else if (t < 384) b = b_v[t - 256];
    else if (t < 480) b = b_qp[t - 384];
    else if (t < 576) b = b_kp[t - 480];
    else b = b_vp[t - 576];
    BCAT[t] = b;
  }
  if (tid < HH) {
    float g = gamma[tid];
    float sp = (g > 20.f) ? g : log1pf(__expf(g));
    PS[tid] = sp * (1.f / 12.f);  // softplus(gamma) * w_C/2, w_C = 1/6
  }
}

// ------------------------------------ K0b: pack projection weights (frag-major)
__global__ void k0_packW(const float* w_q, const float* w_k, const float* w_v,
                         const float* w_qp, const float* w_kp, const float* w_vp,
                         _Float16* PKW) {
  int kb = blockIdx.x, nt = blockIdx.y, lane = threadIdx.x;
  int lo = lane & 15, hi = lane >> 4;
  int c0 = nt * 16;
  const float* W; int ww, col0;
  if (c0 < 128)      { W = w_q;  ww = 128; col0 = c0; }
  else if (c0 < 256) { W = w_k;  ww = 128; col0 = c0 - 128; }
  else if (c0 < 384) { W = w_v;  ww = 128; col0 = c0 - 256; }
  else if (c0 < 480) { W = w_qp; ww = 96;  col0 = c0 - 384; }
  else if (c0 < 576) { W = w_kp; ww = 96;  col0 = c0 - 480; }
  else               { W = w_vp; ww = 96;  col0 = c0 - 576; }
  _Float16* dst = PKW + (((size_t)kb * 42 + nt) * 32 + lane) * 16;
#pragma unroll
  for (int t = 0; t < 16; ++t) {
    int k = kb * 32 + frag_k(t, hi);
    dst[t] = (_Float16)W[k * ww + col0 + lo];
  }
}

// ----------------------------------------------- K0c: pack w_pair (128x64 pad)
__global__ void k0_packWP(const float* w_pair, _Float16* WPB) {
  int kb = blockIdx.x, nt = blockIdx.y, lane = threadIdx.x;
  int lo = lane & 15, hi = lane >> 4;
  _Float16* dst = WPB + (((size_t)kb * 4 + nt) * 32 + lane) * 16;
#pragma unroll
  for (int t = 0; t < 16; ++t) {
    int k = kb * 32 + frag_k(t, hi);
    int n = nt * 16 + lo;
    dst[t] = (k < FDIM && n < PAIRC) ? (_Float16)w_pair[k * PAIRC + n]
                                     : (_Float16)0.f;
  }
}

// ----------------------------------------------------- K0d: pack w_out 224x256
__global__ void k0_packWO(const float* w_out, _Float16* WOB) {
  int kb = blockIdx.x, nt = blockIdx.y, lane = threadIdx.x;
  int lo = lane & 15, hi = lane >> 4;
  _Float16* dst = WOB + (((size_t)kb * 16 + nt) * 32 + lane) * 16;
#pragma unroll
  for (int t = 0; t < 16; ++t) {
    int k = kb * 32 + frag_k(t, hi);
    dst[t] = (_Float16)w_out[k * 256 + nt * 16 + lo];
  }
}

// -------------------------------------------------------- K0e: local -> f16
__global__ void k0_loc16(const float* local, _Float16* LOC16) {
  int i = blockIdx.x, tid = threadIdx.x;
  for (int t = tid; t < DIM; t += blockDim.x)
    LOC16[(size_t)i * DIM + t] = (_Float16)local[(size_t)i * DIM + t];
}

// ------------------------------------------------- K1: projection GEMM (WMMA)
__global__ void k1_proj(const _Float16* __restrict__ LOC16,
                        const _Float16* __restrict__ PKW,
                        const float* __restrict__ BCAT, float* RAW) {
  int i0 = blockIdx.x * 16, nt = blockIdx.y;
  int lane = threadIdx.x;
  int lo = lane & 15, hi = lane >> 4;
  const _Float16* arow = LOC16 + (size_t)(i0 + lo) * DIM;
  v8f acc = {};
#pragma unroll
  for (int kb = 0; kb < DIM / 32; ++kb) {
    v16h a = frag_row(arow + kb * 32, hi);
    v16h b = frag_packed(PKW + (((size_t)kb * 42 + nt) * 32 + lane) * 16);
    acc = wmma_f16(a, b, acc);
  }
  int c0 = nt * 16;
  float bias = BCAT[c0 + lo];
#pragma unroll
  for (int g = 0; g < 8; ++g) {
    int m = g + 8 * hi;
    RAW[(i0 + m) * RAWW + c0 + lo] = acc[g] + bias;
  }
}

// ------------------------- K2: per-token post (LN, +ca, norms, bias scalars)
__global__ void k2_post(const float* RAW, const float* pos,
                        const float* qls, const float* qlo,
                        const float* kls, const float* klo,
                        const float* w_bias,
                        _Float16* QH16, _Float16* KH16,
                        float* QPp, float* KPp, _Float16* QPpad16,
                        _Float16* KP16, _Float16* VAL16,
                        float* DCA, float* NRM) {
  int i = blockIdx.x, lane = threadIdx.x;
  const float* raw = RAW + i * RAWW;
  __shared__ float qp_s[96], kp_s[96];
  float ca0 = pos[i * 15 + 3], ca1 = pos[i * 15 + 4], ca2 = pos[i * 15 + 5];
  if (lane < 3)
    DCA[i * 4 + lane] = 10.f * ((lane == 0) ? ca0 : (lane == 1) ? ca1 : ca2);
  float qsc = qls[lane], qof = qlo[lane], ksc = kls[lane], kof = klo[lane];
#pragma unroll
  for (int h = 0; h < HH; ++h) {
    float xq = raw[h * SS + lane];
    float mu = wave_sum(xq) * (1.f / 32.f);
    float dq = xq - mu;
    float var = wave_sum(dq * dq) * (1.f / 32.f);
    float qn = dq * rsqrtf(var + 1e-5f) * qsc + qof;
    QH16[i * HSX + h * SS + lane] = (_Float16)(qn * 0.17677669529663687f); // sqrt(1/S)
    float xk = raw[HSX + h * SS + lane];
    mu = wave_sum(xk) * (1.f / 32.f);
    float dk = xk - mu;
    var = wave_sum(dk * dk) * (1.f / 32.f);
    KH16[i * HSX + h * SS + lane] =
        (_Float16)(dk * rsqrtf(var + 1e-5f) * ksc + kof);
  }
#pragma unroll
  for (int r = 0; r < 3; ++r) {
    int t = lane + 32 * r;
    float cax = (t % 3 == 0) ? ca0 : (t % 3 == 1) ? ca1 : ca2;
    float qpv = raw[384 + t] + cax;
    float kpv = raw[480 + t] + cax;
    float vpv = raw[576 + t] + cax;
    qp_s[t] = qpv; kp_s[t] = kpv;
    QPp[i * P3 + t] = qpv;
    KPp[i * P3 + t] = kpv;
    KP16[i * P3 + t] = (_Float16)kpv;
    VAL16[((size_t)i * HH + t / 24) * PAIRC + 32 + (t % 24)] = (_Float16)vpv;
  }
#pragma unroll
  for (int r = 0; r < 4; ++r) {
    int t = lane + 32 * r;
    VAL16[((size_t)i * HH + t / SS) * PAIRC + (t % SS)] = (_Float16)raw[256 + t];
  }
  __syncthreads();
  // QPpad16: [h*32 + c], c<24 real, rest zero (fast A-fragment rows for K3)
#pragma unroll
  for (int r = 0; r < 4; ++r) {
    int t = lane + 32 * r;
    int h = t >> 5, c = t & 31;
    QPpad16[(size_t)i * HSX + t] =
        (c < 24) ? (_Float16)qp_s[h * 24 + c] : (_Float16)0.f;
  }
  for (int h = 0; h < HH; ++h) {
    float pq = 0.f, pk = 0.f;
    if (lane < 24) {
      float a = qp_s[h * 24 + lane]; pq = a * a;
      float b = kp_s[h * 24 + lane]; pk = b * b;
    }
    float npq = wave_sum(pq), npk = wave_sum(pk);
    float sq = 0.f, sk = 0.f;
#pragma unroll
    for (int r = 0; r < 3; ++r) {
      int f = lane + 32 * r;
      float wb = w_bias[f * HH + h];
      sq += qp_s[f] * wb;
      sk += kp_s[f] * wb;
    }
    float bq = wave_sum(sq), bk = wave_sum(sk);
    if (lane == 0) {
      float* nr = NRM + ((size_t)i * HH + h) * 4;
      nr[0] = npq; nr[1] = npk; nr[2] = bq; nr[3] = bk;
    }
  }
}

// ------------------- K2b: pack K-side j-tiles for the logits WMMAs (per head)
__global__ void k2b_packJ(const _Float16* KH16, const float* KPp,
                          _Float16* KHB, _Float16* KPB) {
  int jt = blockIdx.x, h = blockIdx.y, lane = threadIdx.x;
  int lo = lane & 15, hi = lane >> 4;
  size_t base = (((size_t)jt * HH + h) * 32 + lane) * 16;
  _Float16* d1 = KHB + base;
  _Float16* d2 = KPB + base;
#pragma unroll
  for (int t = 0; t < 16; ++t) {
    int k = frag_k(t, hi);
    d1[t] = KH16[(jt * 16 + lo) * HSX + h * SS + k];
    d2[t] = (k < 24) ? (_Float16)KPp[(jt * 16 + lo) * P3 + h * 24 + k]
                     : (_Float16)0.f;
  }
}

// -------------------- K2c: pack B = [KP(96) | value_h(56)] for the K=512 GEMM
__global__ void k2c_packB(const _Float16* KP16, const _Float16* VAL16,
                          _Float16* PKB) {
  int kb = blockIdx.x, nt = blockIdx.y, h = blockIdx.z;
  int lane = threadIdx.x;
  int lo = lane & 15, hi = lane >> 4;
  _Float16* dst = PKB + ((((size_t)h * 16 + kb) * 10 + nt) * 32 + lane) * 16;
  int c = nt * 16 + lo;
#pragma unroll
  for (int t = 0; t < 16; ++t) {
    int j = kb * 32 + frag_k(t, hi);
    _Float16 v;
    if (c < P3) v = KP16[j * P3 + c];
    else if (c < P3 + PAIRC) v = VAL16[((size_t)j * HH + h) * PAIRC + (c - P3)];
    else v = (_Float16)0.f;
    dst[t] = v;
  }
}

// ---------------------- K3: logits = (inner + ps*point + bias)/sqrt(3), mask
__global__ void k3_logits(const _Float16* __restrict__ QH16,
                          const _Float16* __restrict__ KHB,
                          const _Float16* __restrict__ QPpad16,
                          const _Float16* __restrict__ KPB,
                          const float* NRM, const float* PS, const float* DCA,
                          const float* w_bias, const float* b_bias,
                          const unsigned char* maskp, const int* batch,
                          float* ATT) {
  int i0 = blockIdx.x * 16, jt = blockIdx.y, j0 = jt * 16;
  int tid = threadIdx.x;
  int h = tid >> 5, lane = tid & 31;
  __shared__ float rbfdot[256 * 4];
  for (int p = tid; p < 256; p += 128) {
    int m = p >> 4, n = p & 15;
    int ii = i0 + m, jj = j0 + n;
    float dx = DCA[ii * 4 + 0] - DCA[jj * 4 + 0];
    float dy = DCA[ii * 4 + 1] - DCA[jj * 4 + 1];
    float dz = DCA[ii * 4 + 2] - DCA[jj * 4 + 2];
    float dist = sqrtf(dx * dx + dy * dy + dz * dz + 1e-6f);
    float a0 = 0, a1 = 0, a2 = 0, a3 = 0;
#pragma unroll
    for (int f = 0; f < 16; ++f) {
      float u = (dist - (1.375f * f + 0.6875f)) * (1.f / 1.375f);
      float r = __expf(-u * u);
      const float* wb = w_bias + (P3 + f) * HH;
      a0 += r * wb[0]; a1 += r * wb[1]; a2 += r * wb[2]; a3 += r * wb[3];
    }
    rbfdot[p * 4 + 0] = a0; rbfdot[p * 4 + 1] = a1;
    rbfdot[p * 4 + 2] = a2; rbfdot[p * 4 + 3] = a3;
  }
  __syncthreads();
  int lo = lane & 15, hi = lane >> 4;
  size_t pkbase = (((size_t)jt * HH + h) * 32 + lane) * 16;
  v16h aq = frag_row(QH16 + (size_t)(i0 + lo) * HSX + h * SS, hi);
  v16h bk = frag_packed(KHB + pkbase);
  v8f inner = {};
  inner = wmma_f16(aq, bk, inner);
  v16h ap = frag_row(QPpad16 + (size_t)(i0 + lo) * HSX + h * SS, hi);
  v16h bp = frag_packed(KPB + pkbase);
  v8f dotpk = {};
  dotpk = wmma_f16(ap, bp, dotpk);
  float ps = PS[h], bb = b_bias[h];
#pragma unroll
  for (int g = 0; g < 8; ++g) {
    int m = g + 8 * hi;
    int ii = i0 + m, jj = j0 + lo;
    const float* ni = NRM + ((size_t)ii * HH + h) * 4;
    const float* nj = NRM + ((size_t)jj * HH + h) * 4;
    float l = inner[g] + ps * (2.f * dotpk[g] - ni[0] - nj[1]) +
              (ni[2] - nj[3] + rbfdot[(m * 16 + lo) * 4 + h] + bb);
    l *= 0.5773502691896258f;
    bool pm = maskp[ii] && maskp[jj] && (batch[ii] == batch[jj]);
    ATT[((size_t)h * NTOK + ii) * NTOK + jj] = pm ? l : -1e9f;
  }
}

// ------------------------------------- K4: softmax over j (f32 + f16 outputs)
__global__ void k4_softmax(float* ATT, _Float16* ATT16, float* RS) {
  int i = blockIdx.x, h = blockIdx.y, tid = threadIdx.x;
  float* row = ATT + ((size_t)h * NTOK + i) * NTOK;
  _Float16* row16 = ATT16 + ((size_t)h * NTOK + i) * NTOK;
  float v0 = row[tid], v1 = row[tid + 256];
  __shared__ float red[256];
  red[tid] = fmaxf(v0, v1);
  __syncthreads();
  for (int o = 128; o > 0; o >>= 1) {
    if (tid < o) red[tid] = fmaxf(red[tid], red[tid + o]);
    __syncthreads();
  }
  float mx = red[0];
  __syncthreads();
  float e0 = (v0 <= -1e8f) ? 0.f : __expf(v0 - mx);
  float e1 = (v1 <= -1e8f) ? 0.f : __expf(v1 - mx);
  red[tid] = e0 + e1;
  __syncthreads();
  for (int o = 128; o > 0; o >>= 1) {
    if (tid < o) red[tid] += red[tid + o];
    __syncthreads();
  }
  float s = red[0];
  float inv = (s > 0.f) ? 1.f / s : 0.f;
  float a0 = e0 * inv, a1 = e1 * inv;
  row[tid] = a0; row[tid + 256] = a1;
  row16[tid] = (_Float16)a0; row16[tid + 256] = (_Float16)a1;
  if (tid == 0) RS[i * HH + h] = (s > 0.f) ? 1.f : 0.f;
}

// ---------------------------- K5: S = A_h @ [KP(96) | value_h(56)]  (K = 512)
__global__ void k5_gemmA(const _Float16* __restrict__ ATT16,
                         const _Float16* __restrict__ PKB, float* SKPV) {
  int i0 = blockIdx.x * 16, nt = blockIdx.y, h = blockIdx.z;
  int lane = threadIdx.x;
  int lo = lane & 15, hi = lane >> 4;
  const _Float16* arow = ATT16 + ((size_t)h * NTOK + i0 + lo) * NTOK;
  v8f acc = {};
#pragma unroll
  for (int kb = 0; kb < NTOK / 32; ++kb) {
    if (kb < 12) __builtin_prefetch(arow + (kb + 4) * 32, 0, 0);
    v16h a = frag_row(arow + kb * 32, hi);
    v16h b =
        frag_packed(PKB + ((((size_t)h * 16 + kb) * 10 + nt) * 32 + lane) * 16);
    acc = wmma_f16(a, b, acc);
  }
#pragma unroll
  for (int g = 0; g < 8; ++g) {
    int m = g + 8 * hi;
    SKPV[((size_t)h * NTOK + i0 + m) * 160 + nt * 16 + lo] = acc[g];
  }
}

// ------------------- K6: non-separable sums: emb gather + RBF (LDS staging)
__global__ void k6_embrbf(const float* ATT, const float* DCA, const int* resi,
                          const int* chain, const float* resi_embedding,
                          float* EMBACC, float* GRBF) {
  __shared__ float E[66 * PAIRC];
  __shared__ float a_s[NTOK];
  __shared__ float d_s[NTOK];
  __shared__ int rd_s[NTOK];
  int tid = threadIdx.x;
  int i = blockIdx.x, h = blockIdx.y;
  for (int t = tid; t < 66 * PAIRC; t += 128) E[t] = resi_embedding[t];
  const float* arow = ATT + ((size_t)h * NTOK + i) * NTOK;
  int ri = resi[i], ci = chain[i];
  float x0 = DCA[i * 4 + 0], y0 = DCA[i * 4 + 1], z0 = DCA[i * 4 + 2];
  for (int j = tid; j < NTOK; j += 128) {
    a_s[j] = arow[j];
    int rd = ri - resi[j];
    rd = (rd < -32) ? -32 : (rd > 32) ? 32 : rd;
    rd += 32;
    if (ci != chain[j]) rd = 65;
    rd_s[j] = rd * PAIRC;
    float dx = x0 - DCA[j * 4 + 0];
    float dy = y0 - DCA[j * 4 + 1];
    float dz = z0 - DCA[j * 4 + 2];
    d_s[j] = sqrtf(dx * dx + dy * dy + dz * dz + 1e-6f);
  }
  __syncthreads();
  if (tid < PAIRC) {
    int c = tid;
    float acc = 0.f;
    for (int j = 0; j < NTOK; ++j) acc += a_s[j] * E[rd_s[j] + c];
    EMBACC[((size_t)i * HH + h) * PAIRC + c] = acc;
  } else if (tid >= 64 && tid < 80) {
    int f = tid - 64;
    float cen = 1.375f * f + 0.6875f;
    float acc = 0.f;
    for (int j = 0; j < NTOK; ++j) {
      float u = (d_s[j] - cen) * (1.f / 1.375f);
      acc += a_s[j] * __expf(-u * u);
    }
    GRBF[((size_t)i * HH + h) * 16 + f] = acc;
  }
}

// ------------------------- K7: assemble G (rowsum*qp - A@KP | RBF) and resid
__global__ void k7_buildG(const float* RS, const float* QPp, const float* SKPV,
                          const float* GRBF, const float* EMBACC,
                          _Float16* G16, float* RESID) {
  int r = blockIdx.x, f = threadIdx.x;
  int i = r >> 2, h = r & 3;
  float g;
  if (f < P3)
    g = RS[i * HH + h] * QPp[i * P3 + f] - SKPV[((size_t)h * NTOK + i) * 160 + f];
  else if (f < FDIM)
    g = GRBF[((size_t)i * HH + h) * 16 + (f - P3)];
  else
    g = 0.f;
  G16[(size_t)r * 128 + f] = (_Float16)g;
  if (f < PAIRC)
    RESID[(size_t)r * PAIRC + f] = EMBACC[((size_t)i * HH + h) * PAIRC + f] +
                                   SKPV[((size_t)h * NTOK + i) * 160 + P3 + f];
}

// ------------------------------------------------ K8: RES = G @ w_pair + resid
__global__ void k8_gemmPair(const _Float16* __restrict__ G16,
                            const _Float16* __restrict__ WPB,
                            const float* RESID, _Float16* RES16) {
  int r0 = blockIdx.x * 16, nt = blockIdx.y;
  int lane = threadIdx.x;
  int lo = lane & 15, hi = lane >> 4;
  const _Float16* grow = G16 + (size_t)(r0 + lo) * 128;
  v8f acc = {};
#pragma unroll
  for (int kb = 0; kb < 4; ++kb) {
    v16h a = frag_row(grow + kb * 32, hi);
    v16h b = frag_packed(WPB + (((size_t)kb * 4 + nt) * 32 + lane) * 16);
    acc = wmma_f16(a, b, acc);
  }
#pragma unroll
  for (int g = 0; g < 8; ++g) {
    int m = g + 8 * hi;
    int r = r0 + m, c = nt * 16 + lo;
    if (c < PAIRC) {
      int i = r >> 2, h = r & 3;
      float v = acc[g] + RESID[(size_t)r * PAIRC + c];
      RES16[(size_t)i * 224 + h * PAIRC + c] = (_Float16)v;
    }
  }
}

// ------------------------------------------------- K9: out = RES @ w_out
__global__ void k9_out(const _Float16* __restrict__ RES16,
                       const _Float16* __restrict__ WOB, float* out) {
  int i0 = blockIdx.x * 16, nt = blockIdx.y;
  int lane = threadIdx.x;
  int lo = lane & 15, hi = lane >> 4;
  const _Float16* rrow = RES16 + (size_t)(i0 + lo) * 224;
  v8f acc = {};
#pragma unroll
  for (int kb = 0; kb < 7; ++kb) {
    v16h a = frag_row(rrow + kb * 32, hi);
    v16h b = frag_packed(WOB + (((size_t)kb * 16 + nt) * 32 + lane) * 16);
    acc = wmma_f16(a, b, acc);
  }
#pragma unroll
  for (int g = 0; g < 8; ++g) {
    int m = g + 8 * hi;
    out[(size_t)(i0 + m) * 256 + nt * 16 + lo] = acc[g];
  }
}

extern "C" void kernel_launch(void* const* d_in, const int* in_sizes, int n_in,
                              void* d_out, int out_size, void* d_ws,
                              size_t ws_size, hipStream_t stream) {
  (void)in_sizes; (void)n_in; (void)out_size; (void)ws_size;
  const float* local = (const float*)d_in[0];
  const float* pos = (const float*)d_in[1];
  const int* resi = (const int*)d_in[2];
  const int* chain = (const int*)d_in[3];
  const int* batch = (const int*)d_in[4];
  const unsigned char* maskp = (const unsigned char*)d_in[5];
  const float* qls = (const float*)d_in[6];
  const float* qlo = (const float*)d_in[7];
  const float* kls = (const float*)d_in[8];
  const float* klo = (const float*)d_in[9];
  const float* w_q = (const float*)d_in[10];  const float* b_q = (const float*)d_in[11];
  const float* w_k = (const float*)d_in[12];  const float* b_k = (const float*)d_in[13];
  const float* w_v = (const float*)d_in[14];  const float* b_v = (const float*)d_in[15];
  const float* w_qp = (const float*)d_in[16]; const float* b_qp = (const float*)d_in[17];
  const float* w_kp = (const float*)d_in[18]; const float* b_kp = (const float*)d_in[19];
  const float* w_vp = (const float*)d_in[20]; const float* b_vp = (const float*)d_in[21];
  const float* remb = (const float*)d_in[22];
  const float* w_pair = (const float*)d_in[23];
  const float* w_bias = (const float*)d_in[24];
  const float* b_bias = (const float*)d_in[25];
  const float* gamma = (const float*)d_in[26];
  const float* w_out = (const float*)d_in[27];

  char* ws = (char*)d_ws;
  size_t off = 0;
  auto alloc = [&](size_t bytes) -> void* {
    off = (off + 255) & ~(size_t)255;
    void* p = ws + off;
    off += bytes;
    return p;
  };
  float* RAW = (float*)alloc((size_t)NTOK * RAWW * 4);
  _Float16* LOC16 = (_Float16*)alloc((size_t)NTOK * DIM * 2);
  _Float16* PKW = (_Float16*)alloc((size_t)8 * 42 * 32 * 16 * 2);
  float* BCAT = (float*)alloc(RAWW * 4);
  _Float16* QH16 = (_Float16*)alloc((size_t)NTOK * HSX * 2);
  _Float16* KH16 = (_Float16*)alloc((size_t)NTOK * HSX * 2);
  float* QPp = (float*)alloc((size_t)NTOK * P3 * 4);
  float* KPp = (float*)alloc((size_t)NTOK * P3 * 4);
  _Float16* QPpad16 = (_Float16*)alloc((size_t)NTOK * HSX * 2);
  _Float16* KP16 = (_Float16*)alloc((size_t)NTOK * P3 * 2);
  _Float16* VAL16 = (_Float16*)alloc((size_t)NTOK * HH * PAIRC * 2);
  _Float16* KHB = (_Float16*)alloc((size_t)32 * HH * 32 * 16 * 2);
  _Float16* KPB = (_Float16*)alloc((size_t)32 * HH * 32 * 16 * 2);
  _Float16* PKB = (_Float16*)alloc((size_t)HH * 16 * 10 * 32 * 16 * 2);
  float* DCA = (float*)alloc((size_t)NTOK * 4 * 4);
  float* NRM = (float*)alloc((size_t)NTOK * HH * 4 * 4);
  float* PS = (float*)alloc(256);
  float* ATT = (float*)alloc((size_t)HH * NTOK * NTOK * 4);
  _Float16* ATT16 = (_Float16*)alloc((size_t)HH * NTOK * NTOK * 2);
  float* RS = (float*)alloc((size_t)NTOK * HH * 4);
  float* SKPV = (float*)alloc((size_t)HH * NTOK * 160 * 4);
  float* EMBACC = (float*)alloc((size_t)NTOK * HH * PAIRC * 4);
  float* GRBF = (float*)alloc((size_t)NTOK * HH * 16 * 4);
  _Float16* G16 = (_Float16*)alloc((size_t)NTOK * HH * 128 * 2);
  float* RESID = (float*)alloc((size_t)NTOK * HH * PAIRC * 4);
  _Float16* RES16 = (_Float16*)alloc((size_t)NTOK * 224 * 2);
  _Float16* WPB = (_Float16*)alloc((size_t)4 * 4 * 32 * 16 * 2);
  _Float16* WOB = (_Float16*)alloc((size_t)7 * 16 * 32 * 16 * 2);

  k0_misc<<<dim3(1), dim3(256), 0, stream>>>(b_q, b_k, b_v, b_qp, b_kp, b_vp,
                                             gamma, BCAT, PS);
  k0_packW<<<dim3(8, 42), dim3(32), 0, stream>>>(w_q, w_k, w_v, w_qp, w_kp,
                                                 w_vp, PKW);
  k0_packWP<<<dim3(4, 4), dim3(32), 0, stream>>>(w_pair, WPB);
  k0_packWO<<<dim3(7, 16), dim3(32), 0, stream>>>(w_out, WOB);
  k0_loc16<<<dim3(NTOK), dim3(64), 0, stream>>>(local, LOC16);
  k1_proj<<<dim3(32, 42), dim3(32), 0, stream>>>(LOC16, PKW, BCAT, RAW);
  k2_post<<<dim3(NTOK), dim3(32), 0, stream>>>(RAW, pos, qls, qlo, kls, klo,
                                               w_bias, QH16, KH16, QPp, KPp,
                                               QPpad16, KP16, VAL16, DCA, NRM);
  k2b_packJ<<<dim3(32, HH), dim3(32), 0, stream>>>(KH16, KPp, KHB, KPB);
  k2c_packB<<<dim3(16, 10, HH), dim3(32), 0, stream>>>(KP16, VAL16, PKB);
  k3_logits<<<dim3(32, 32), dim3(128), 0, stream>>>(QH16, KHB, QPpad16, KPB,
                                                    NRM, PS, DCA, w_bias,
                                                    b_bias, maskp, batch, ATT);
  k4_softmax<<<dim3(NTOK, HH), dim3(256), 0, stream>>>(ATT, ATT16, RS);
  k5_gemmA<<<dim3(32, 10, HH), dim3(32), 0, stream>>>(ATT16, PKB, SKPV);
  k6_embrbf<<<dim3(NTOK, HH), dim3(128), 0, stream>>>(ATT, DCA, resi, chain,
                                                      remb, EMBACC, GRBF);
  k7_buildG<<<dim3(NTOK * HH), dim3(128), 0, stream>>>(RS, QPp, SKPV, GRBF,
                                                       EMBACC, G16, RESID);
  k8_gemmPair<<<dim3(128, 4), dim3(32), 0, stream>>>(G16, WPB, RESID, RES16);
  k9_out<<<dim3(32, 16), dim3(32), 0, stream>>>(RES16, WOB, (float*)d_out);
}
